// SlotAttention_67284957659359
// MI455X (gfx1250) — compile-verified, matrix operand order checked by reading, not run on hardware
//
#include <hip/hip_runtime.h>
#include <math.h>

typedef __bf16 bf16_t;
typedef __attribute__((ext_vector_type(16))) __bf16 v16bf;
typedef __attribute__((ext_vector_type(8)))  __bf16 v8bf;
typedef __attribute__((ext_vector_type(8)))  float  v8f;

union U16 { v16bf v; v8bf h[2]; };
union U8B { v8bf v; bf16_t e[8]; };

#define DIM_      256
#define NTOK_     1024
#define BATCH_    128
#define ROWS_BLK  64
#define XS_STRIDE 260   // fp32 elems; 1040B row, multiple of 16B, de-conflicts banks
#define XT_STRIDE 264   // bf16 elems; 528B row, multiple of 16B
#define AB_STRIDE 1040  // bf16 elems; 2080B row, multiple of 16B

// ---------------------------------------------------------------------------
// Weight prep: fp32 W[kin][nout] (row-major)  ->  bf16 WT[nout][kin]
// ---------------------------------------------------------------------------
__global__ __launch_bounds__(256) void transpose_bf16(const float* __restrict__ src,
                                                      bf16_t* __restrict__ dst,
                                                      int kin, int nout) {
    int idx = blockIdx.x * 256 + threadIdx.x;
    if (idx < kin * nout) {
        int n = idx / kin;
        int k = idx - n * kin;
        dst[idx] = (bf16_t)src[(size_t)k * nout + n];
    }
}

// ---------------------------------------------------------------------------
// Kernel 1: x = LN(tokens); k = x@Wk+bk (store bf16 row-major (n,d));
//           v = x@Wv+bv (store bf16 transposed (d,n))
// One block = 64 token rows. Token tile staged into LDS with
// GLOBAL_LOAD_ASYNC_TO_LDS_B128 (ASYNCcnt). 8 waves; tasks = {K,V} x 16
// n-tiles; each wave register-blocks 4 m-tiles (B-fragment reuse x4).
// ---------------------------------------------------------------------------
__global__ __launch_bounds__(256) void ln_kv_kernel(
    const float* __restrict__ tokens,
    const float* __restrict__ lng, const float* __restrict__ lnb,
    const float* __restrict__ bk,  const float* __restrict__ bv,
    const bf16_t* __restrict__ WkT, const bf16_t* __restrict__ WvT,
    bf16_t* __restrict__ kout, bf16_t* __restrict__ vT)
{
    __shared__ float  xs[ROWS_BLK][XS_STRIDE];  // fp32 token stage (~66.5 KB)
    __shared__ bf16_t xt[ROWS_BLK][XT_STRIDE];  // bf16 LN'd A tile (~33.8 KB)

    const int tid = threadIdx.x;
    const size_t row0 = (size_t)blockIdx.x * ROWS_BLK;

    // ---- async copy: 64 rows x 256 fp32 -> LDS, 16B per lane per issue ----
    #pragma unroll
    for (int j = 0; j < 16; ++j) {
        const int c    = tid + 256 * j;          // chunk id 0..4095
        const int row  = c >> 6;                 // 64 chunks per row
        const int col4 = (c & 63) << 2;          // fp32 column (x4 floats)
        const float* g = tokens + (row0 + row) * DIM_ + col4;
        const unsigned lptr = (unsigned)(size_t)&xs[row][col4];
        asm volatile("global_load_async_to_lds_b128 %0, %1, off"
                     :: "v"(lptr), "v"(g) : "memory");
    }
    asm volatile("s_wait_asynccnt 0x0" ::: "memory");
    __syncthreads();

    // ---- LayerNorm from LDS: 4 lanes per row, 64 elems per lane ----
    {
        const int r = tid >> 2;
        const int q = tid & 3;
        const float* srcl = &xs[r][q * 64];
        float s = 0.f, ss = 0.f;
        #pragma unroll
        for (int i = 0; i < 64; ++i) { float x = srcl[i]; s += x; ss += x * x; }
        s  += __shfl_xor(s, 1, 32);  ss += __shfl_xor(ss, 1, 32);
        s  += __shfl_xor(s, 2, 32);  ss += __shfl_xor(ss, 2, 32);
        const float mean = s * (1.f / 256.f);
        const float var  = ss * (1.f / 256.f) - mean * mean;
        const float rstd = rsqrtf(var + 1e-6f);
        bf16_t* dst = &xt[r][q * 64];
        #pragma unroll
        for (int i = 0; i < 64; ++i) {
            float xn = (srcl[i] - mean) * rstd * lng[q * 64 + i] + lnb[q * 64 + i];
            dst[i] = (bf16_t)xn;
        }
    }
    __syncthreads();

    // ---- GEMM phase ----
    const int w    = tid >> 5;
    const int lane = tid & 31;
    const int half = lane >> 4;

    for (int t = 0; t < 4; ++t) {
        const int task = w + t * 8;      // 0..31
        const int mat  = task >> 4;      // 0 -> K, 1 -> V
        const int nt   = task & 15;
        const bf16_t* WT  = mat ? WvT : WkT;
        const float* bias = mat ? bv  : bk;
        const int ncol = nt * 16 + (lane & 15);
        const float bi = bias[ncol];
        v8f acc[4];
        #pragma unroll
        for (int mt = 0; mt < 4; ++mt) acc[mt] = (v8f){bi, bi, bi, bi, bi, bi, bi, bi};

        const bf16_t* brow = WT + (size_t)ncol * DIM_;
        for (int ks = 0; ks < 8; ++ks) {
            const int k0 = ks * 32 + (half << 3);
            U16 ub;
            ub.h[0] = *(const v8bf*)(brow + k0);
            ub.h[1] = *(const v8bf*)(brow + k0 + 16);
            #pragma unroll
            for (int mt = 0; mt < 4; ++mt) {
                const int m = mt * 16 + (lane & 15);
                U16 ua;
                ua.h[0] = *(const v8bf*)&xt[m][k0];
                ua.h[1] = *(const v8bf*)&xt[m][k0 + 16];
                acc[mt] = __builtin_amdgcn_wmma_f32_16x16x32_bf16(
                    false, ua.v, false, ub.v, (short)0, acc[mt], false, false);
            }
        }
        // store
        #pragma unroll
        for (int mt = 0; mt < 4; ++mt) {
            if (mat == 0) {
                #pragma unroll
                for (int r = 0; r < 8; ++r) {
                    const size_t grow = row0 + mt * 16 + r + (half << 3);
                    kout[grow * DIM_ + ncol] = (bf16_t)acc[mt][r];
                }
            } else {
                // 8 accumulator rows = 8 consecutive tokens at fixed d -> one 16B store
                U8B pk;
                #pragma unroll
                for (int r = 0; r < 8; ++r) pk.e[r] = (bf16_t)acc[mt][r];
                const size_t grow0 = row0 + mt * 16 + (half << 3);
                const size_t bb = grow0 >> 10;
                const size_t nn = grow0 & 1023;
                *(v8bf*)&vT[bb * (size_t)(DIM_ * NTOK_) + (size_t)ncol * NTOK_ + nn] = pk.v;
            }
        }
    }
}

// ---------------------------------------------------------------------------
// Generic M=8 (padded to 16) GEMM for the slot kernel.
// A: bf16 LDS tile [8][AB_STRIDE] (rows 8..15 synthesized zero)
// B: bf16 global, layout [nout][kdim] (per-row contiguous K)
// out: fp32 LDS, row stride ostride. mode 0=store 1=accumulate 2=gelu-store
// ---------------------------------------------------------------------------
__device__ __forceinline__ void gemm8(
    const bf16_t (&A)[8][AB_STRIDE], int kdim,
    const bf16_t* __restrict__ Bw,
    const float* __restrict__ bias, float scale,
    float* __restrict__ out, int ostride, int nout, int mode)
{
    const int tid  = threadIdx.x;
    const int w    = tid >> 5;
    const int lane = tid & 31;
    const int m    = lane & 15;
    const int half = lane >> 4;
    const int ntiles = nout >> 4;
    const int ksteps = kdim >> 5;

    const __bf16 zb = (__bf16)0.0f;
    const v16bf vz = {zb, zb, zb, zb, zb, zb, zb, zb, zb, zb, zb, zb, zb, zb, zb, zb};

    for (int nt = w; nt < ntiles; nt += 8) {
        const int ncol = (nt << 4) + m;
        const float bi = bias ? bias[ncol] : 0.0f;
        v8f acc = (v8f){bi, bi, bi, bi, bi, bi, bi, bi};
        const bf16_t* brow = Bw + (size_t)ncol * kdim;
        for (int ks = 0; ks < ksteps; ++ks) {
            const int k0 = (ks << 5) + (half << 3);
            U16 ua;
            if (m < 8) {
                ua.h[0] = *(const v8bf*)&A[m][k0];
                ua.h[1] = *(const v8bf*)&A[m][k0 + 16];
            } else {
                ua.v = vz;
            }
            U16 ub;
            ub.h[0] = *(const v8bf*)(brow + k0);
            ub.h[1] = *(const v8bf*)(brow + k0 + 16);
            acc = __builtin_amdgcn_wmma_f32_16x16x32_bf16(
                false, ua.v, false, ub.v, (short)0, acc, false, false);
        }
        if (half == 0) {
            #pragma unroll
            for (int r = 0; r < 8; ++r) {
                float vv = acc[r] * scale;
                if (mode == 2) vv = 0.5f * vv * (1.0f + erff(vv * 0.70710678118654752f));
                if (mode == 1) out[r * ostride + ncol] += vv;
                else           out[r * ostride + ncol]  = vv;
            }
        }
    }
}

__device__ __forceinline__ void ln8_to_abf(const float (&src)[8][256],
                                           const float* __restrict__ g,
                                           const float* __restrict__ bb,
                                           bf16_t (&dst)[8][AB_STRIDE])
{
    const int w = threadIdx.x >> 5, lane = threadIdx.x & 31;
    float vals[8]; float s = 0.f, ss = 0.f;
    #pragma unroll
    for (int i = 0; i < 8; ++i) { float x = src[w][lane * 8 + i]; vals[i] = x; s += x; ss += x * x; }
    #pragma unroll
    for (int off = 1; off < 32; off <<= 1) { s += __shfl_xor(s, off, 32); ss += __shfl_xor(ss, off, 32); }
    const float mean = s * (1.f / 256.f);
    const float var  = ss * (1.f / 256.f) - mean * mean;
    const float rstd = rsqrtf(var + 1e-6f);
    #pragma unroll
    for (int i = 0; i < 8; ++i) {
        const int c = lane * 8 + i;
        dst[w][c] = (bf16_t)((vals[i] - mean) * rstd * g[c] + bb[c]);
    }
}

__device__ __forceinline__ void copy_to_abf(const float* __restrict__ src, int sstride,
                                            int cols, bf16_t (&dst)[8][AB_STRIDE])
{
    for (int idx = threadIdx.x; idx < 8 * cols; idx += 256) {
        const int r = idx / cols;
        const int c = idx - r * cols;
        dst[r][c] = (bf16_t)src[r * sstride + c];
    }
}

// ---------------------------------------------------------------------------
// Kernel 2: 3 slot-attention iterations. One block per batch.
// ---------------------------------------------------------------------------
__global__ __launch_bounds__(256) void slot_iter_kernel(
    const float* __restrict__ init_slots,
    const float* __restrict__ ln_s_g, const float* __restrict__ ln_s_b,
    const float* __restrict__ bq,  const bf16_t* __restrict__ WqT,
    const float* __restrict__ b_ih, const bf16_t* __restrict__ WihT,
    const float* __restrict__ b_hh, const bf16_t* __restrict__ WhhT,
    const float* __restrict__ ln_m_g, const float* __restrict__ ln_m_b,
    const float* __restrict__ b1, const bf16_t* __restrict__ W1T,
    const float* __restrict__ b2, const bf16_t* __restrict__ W2T,
    const bf16_t* __restrict__ kbf, const bf16_t* __restrict__ vT,
    float* __restrict__ out)
{
    __shared__ float  s_slots[8][256];       //  8 KB
    __shared__ bf16_t s_abf[8][AB_STRIDE];   // 16.6 KB
    __shared__ float  s_logits[8][1024];     // 32 KB
    __shared__ float  s_big[8][2048];        // 64 KB
    __shared__ float  s_upd[8][256];         //  8 KB
    __shared__ float  s_rs[8];

    const int tid = threadIdx.x;
    const int b   = blockIdx.x;
    const bf16_t* kb = kbf + (size_t)b * NTOK_ * DIM_;   // k[n][d]
    const bf16_t* vb = vT  + (size_t)b * DIM_ * NTOK_;   // vT[d][n]
    float* slots_out = out + (size_t)b * 8 * DIM_;
    float* masks_out = out + (size_t)BATCH_ * 8 * DIM_ + (size_t)b * 8 * NTOK_;

    for (int idx = tid; idx < 2048; idx += 256)
        s_slots[idx >> 8][idx & 255] = init_slots[(size_t)b * 2048 + idx];
    __syncthreads();

    for (int iter = 0; iter < 3; ++iter) {
        const bool last = (iter == 2);

        // q = LN(slots) @ Wq + bq
        ln8_to_abf(s_slots, ln_s_g, ln_s_b, s_abf);
        __syncthreads();
        gemm8(s_abf, 256, WqT, bq, 1.0f, &s_upd[0][0], 256, 256, 0);
        __syncthreads();
        copy_to_abf(&s_upd[0][0], 256, 256, s_abf);
        __syncthreads();

        // logits = q @ k^T * (1/sqrt(D))
        gemm8(s_abf, 256, kb, nullptr, 0.0625f, &s_logits[0][0], 1024, 1024, 0);
        __syncthreads();

        // softmax over slot axis + masks + renormalized attn
        if (tid < 8) s_rs[tid] = 0.0f;
        __syncthreads();
        float partial[8];
        #pragma unroll
        for (int k = 0; k < 8; ++k) partial[k] = 0.0f;
        for (int c = tid; c < 1024; c += 256) {
            float mx = s_logits[0][c];
            #pragma unroll
            for (int k = 1; k < 8; ++k) mx = fmaxf(mx, s_logits[k][c]);
            float e[8]; float sum = 0.0f;
            #pragma unroll
            for (int k = 0; k < 8; ++k) { e[k] = __expf(s_logits[k][c] - mx); sum += e[k]; }
            const float inv = 1.0f / sum;
            #pragma unroll
            for (int k = 0; k < 8; ++k) {
                const float mk = e[k] * inv;
                s_logits[k][c] = mk;
                if (last) masks_out[k * 1024 + c] = mk;
                partial[k] += mk + 1e-6f;
            }
        }
        #pragma unroll
        for (int k = 0; k < 8; ++k) atomicAdd(&s_rs[k], partial[k]);
        __syncthreads();
        for (int c = tid; c < 1024; c += 256) {
            #pragma unroll
            for (int k = 0; k < 8; ++k)
                s_abf[k][c] = (bf16_t)((s_logits[k][c] + 1e-6f) / (s_rs[k] + 1e-6f));
        }
        __syncthreads();

        // updates = attn @ v   (B = vT[d][n], K = N = 1024)
        gemm8(s_abf, 1024, vb, nullptr, 1.0f, &s_upd[0][0], 256, 256, 0);
        __syncthreads();

        // GRU: gi = updates@W_ih + b_ih ; gh = slots@W_hh + b_hh
        copy_to_abf(&s_upd[0][0], 256, 256, s_abf);
        __syncthreads();
        gemm8(s_abf, 256, WihT, b_ih, 1.0f, &s_big[0][0], 2048, 768, 0);
        __syncthreads();
        copy_to_abf(&s_slots[0][0], 256, 256, s_abf);
        __syncthreads();
        gemm8(s_abf, 256, WhhT, b_hh, 1.0f, &s_big[0][768], 2048, 768, 0);
        __syncthreads();
        {
            const int k  = tid >> 5;
            const int n0 = (tid & 31) * 8;
            #pragma unroll
            for (int j = 0; j < 8; ++j) {
                const int n = n0 + j;
                const float ir = s_big[k][n],        iz = s_big[k][256 + n],  inn = s_big[k][512 + n];
                const float hr = s_big[k][768 + n],  hz = s_big[k][1024 + n], hn  = s_big[k][1280 + n];
                const float r  = 1.0f / (1.0f + __expf(-(ir + hr)));
                const float z  = 1.0f / (1.0f + __expf(-(iz + hz)));
                const float nn = tanhf(inn + r * hn);
                s_slots[k][n] = (1.0f - z) * nn + z * s_slots[k][n];
            }
        }
        __syncthreads();

        // MLP: slots += gelu(LN(slots)@W1 + b1) @ W2 + b2
        ln8_to_abf(s_slots, ln_m_g, ln_m_b, s_abf);
        __syncthreads();
        gemm8(s_abf, 256, W1T, b1, 1.0f, &s_big[0][0], 2048, 1024, 2);
        __syncthreads();
        copy_to_abf(&s_big[0][0], 2048, 1024, s_abf);
        __syncthreads();
        gemm8(s_abf, 1024, W2T, b2, 1.0f, &s_slots[0][0], 256, 256, 1);
        __syncthreads();
    }

    for (int idx = tid; idx < 2048; idx += 256)
        slots_out[idx] = s_slots[idx >> 8][idx & 255];
}

// ---------------------------------------------------------------------------
extern "C" void kernel_launch(void* const* d_in, const int* in_sizes, int n_in,
                              void* d_out, int out_size, void* d_ws, size_t ws_size,
                              hipStream_t stream) {
    (void)in_sizes; (void)n_in; (void)out_size; (void)ws_size;

    const float* tokens     = (const float*)d_in[0];
    const float* init_slots = (const float*)d_in[1];
    const float* ln_in_g = (const float*)d_in[2];
    const float* ln_in_b = (const float*)d_in[3];
    const float* Wk = (const float*)d_in[4];  const float* bk = (const float*)d_in[5];
    const float* Wv = (const float*)d_in[6];  const float* bv = (const float*)d_in[7];
    const float* ln_s_g = (const float*)d_in[8];
    const float* ln_s_b = (const float*)d_in[9];
    const float* Wq = (const float*)d_in[10]; const float* bq = (const float*)d_in[11];
    const float* W_ih = (const float*)d_in[12]; const float* b_ih = (const float*)d_in[13];
    const float* W_hh = (const float*)d_in[14]; const float* b_hh = (const float*)d_in[15];
    const float* ln_m_g = (const float*)d_in[16];
    const float* ln_m_b = (const float*)d_in[17];
    const float* W1 = (const float*)d_in[18]; const float* b1 = (const float*)d_in[19];
    const float* W2 = (const float*)d_in[20]; const float* b2 = (const float*)d_in[21];

    // Workspace carve
    size_t off = 0;
    char* base = (char*)d_ws;
    auto carve = [&](size_t bytes) -> void* {
        void* p = base + off;
        off += (bytes + 255) & ~(size_t)255;
        return p;
    };
    bf16_t* kbf  = (bf16_t*)carve((size_t)BATCH_ * NTOK_ * DIM_ * 2);
    bf16_t* vTw  = (bf16_t*)carve((size_t)BATCH_ * NTOK_ * DIM_ * 2);
    bf16_t* WkT  = (bf16_t*)carve(256 * 256 * 2);
    bf16_t* WvT  = (bf16_t*)carve(256 * 256 * 2);
    bf16_t* WqT  = (bf16_t*)carve(256 * 256 * 2);
    bf16_t* WihT = (bf16_t*)carve(768 * 256 * 2);
    bf16_t* WhhT = (bf16_t*)carve(768 * 256 * 2);
    bf16_t* W1T  = (bf16_t*)carve(1024 * 256 * 2);
    bf16_t* W2T  = (bf16_t*)carve(256 * 1024 * 2);

    auto T = [&](const float* src, bf16_t* dst, int kin, int nout) {
        int n = kin * nout;
        transpose_bf16<<<(n + 255) / 256, 256, 0, stream>>>(src, dst, kin, nout);
    };
    T(Wk,   WkT,  256, 256);
    T(Wv,   WvT,  256, 256);
    T(Wq,   WqT,  256, 256);
    T(W_ih, WihT, 256, 768);
    T(W_hh, WhhT, 256, 768);
    T(W1,   W1T,  256, 1024);
    T(W2,   W2T,  1024, 256);

    ln_kv_kernel<<<(BATCH_ * NTOK_) / ROWS_BLK, 256, 0, stream>>>(
        tokens, ln_in_g, ln_in_b, bk, bv, WkT, WvT, kbf, vTw);

    slot_iter_kernel<<<BATCH_, 256, 0, stream>>>(
        init_slots, ln_s_g, ln_s_b, bq, WqT, b_ih, WihT, b_hh, WhhT,
        ln_m_g, ln_m_b, b1, W1T, b2, W2T, kbf, vTw, (float*)d_out);
}
